// GruCopyingDecoder_50302656971018
// MI455X (gfx1250) — compile-verified, matrix operand order checked by reading, not run on hardware
//
#include <hip/hip_runtime.h>
#include <hip/hip_bf16.h>
#include <math.h>
#include <stdint.h>

// ---------------- problem constants ----------------
#define BB   128
#define TT   33
#define LL   32
#define VV   25000
#define EE   256
#define HH   256
#define DD   512
#define NN   8192
#define CC   16384
#define BL   (BB*LL)            // 4096
#define SEG  64                 // N/B memories per sample
#define NCHUNK 196              // ceil(25000/128)

typedef __attribute__((ext_vector_type(16))) __bf16 v16bf;
typedef __attribute__((ext_vector_type(8)))  float  v8f;

#define NEG_INF (-__builtin_inff())
#define AS3 __attribute__((address_space(3)))

// ---------------- CDNA5 async global->LDS copy (ASYNCcnt-tracked) ----------------
// global_load_async_to_lds_b128: VDST = VGPR holding LDS byte address,
// VADDR = 64-bit global address. Tracked with ASYNCcnt (in-order completion).
__device__ __forceinline__ void async_cp16(__bf16* lds, const __bf16* g) {
    unsigned l = (unsigned)(uintptr_t)(AS3 __bf16*)lds;
    asm volatile("global_load_async_to_lds_b128 %0, %1, off"
                 :: "v"(l), "v"(g) : "memory");
}

template <int N>
__device__ __forceinline__ void wait_async() {
#if __has_builtin(__builtin_amdgcn_s_wait_asynccnt)
    __builtin_amdgcn_s_wait_asynccnt(N);
#else
    asm volatile("s_wait_asynccnt %0" :: "i"(N) : "memory");
#endif
}

// ---------------- WMMA helpers ----------------
// CDNA5 16-bit A/B fragment: lane holds row (A) / col (B) = lane%16,
// k-values: elements 0..7 -> k = k0 + 8*(lane>=16) + e ; elements 8..15 -> +16.
// Both 8-element groups are contiguous bf16 in row-major [row][k] storage,
// so a fragment = two 16-byte LDS loads.
__device__ __forceinline__ v16bf ldfrag(const __bf16* p, int row0, int ld, int k0, int lane) {
    int r  = row0 + (lane & 15);
    int kb = k0 + ((lane >> 4) << 3);
    const __bf16* q = p + r * ld + kb;
    union { v16bf v; uint4 u[2]; } x;
    x.u[0] = *(const uint4*)(q);
    x.u[1] = *(const uint4*)(q + 16);
    return x.v;
}

__device__ __forceinline__ v8f wmma_bf16(v16bf a, v16bf b, v8f c) {
    return __builtin_amdgcn_wmma_f32_16x16x32_bf16(false, a, false, b, (short)0, c, false, false);
}

// C/D tile layout: lane holds col = lane%16, rows j + 8*(lane>=16), j = VGPR 0..7
__device__ __forceinline__ void store_tile(v8f c, int row0, int col0, int ldc,
                                           float* Cf, __bf16* Cb, const float* bias, int lane) {
    int cn  = lane & 15;
    int rb  = (lane >> 4) << 3;
    int col = col0 + cn;
    float bv = bias ? bias[col] : 0.f;
#pragma unroll
    for (int j = 0; j < 8; ++j) {
        float v = c[j] + bv;
        long idx = (long)(row0 + rb + j) * ldc + col;
        if (Cf) Cf[idx] = v;
        if (Cb) Cb[idx] = (__bf16)v;
    }
}

// ---------------- elementwise converts ----------------
__global__ void k_cvt_bf16(const float* __restrict__ in, __bf16* __restrict__ out, int n) {
    int i = blockIdx.x * blockDim.x + threadIdx.x;
    if (i < n) out[i] = (__bf16)in[i];
}

// out[c*R + r] = in[r*C + c]   (transpose-convert, for hidden_to_vocab)
__global__ void k_cvt_t_bf16(const float* __restrict__ in, __bf16* __restrict__ out, int R, int Cc) {
    int i = blockIdx.x * blockDim.x + threadIdx.x;
    if (i < R * Cc) {
        int r = i / Cc, c = i % Cc;
        out[c * R + r] = (__bf16)in[i];
    }
}

// X[r,e] = emb[tok_in[r], e]  in bf16, r = b*L + l
__global__ __launch_bounds__(256) void k_xgather(const float* __restrict__ emb,
                                                 const int* __restrict__ tok,
                                                 __bf16* __restrict__ xb) {
    int r = blockIdx.x, e = threadIdx.x;
    int b = r >> 5, l = r & 31;
    int t = tok[b * TT + l];           // tok_in = tokens[:, :-1]
    xb[r * EE + e] = (__bf16)emb[t * EE + e];
}

// ---------------- generic WMMA GEMM: C[MxN] = A[MxK] @ Bt[NxK]^T (+bias) ----------------
// Double-buffered async global->LDS staging: 4 async b128 per thread per stage.
__global__ __launch_bounds__(256) void k_gemm_bf16(const __bf16* __restrict__ A,
                                                   const __bf16* __restrict__ Bt,
                                                   float* Cf, __bf16* Cb,
                                                   const float* bias,
                                                   int M, int N, int K) {
    __shared__ __bf16 As[2][64 * 64];
    __shared__ __bf16 Bs[2][64 * 64];
    int bm = blockIdx.x, bn = blockIdx.y;
    int tid = threadIdx.x, lane = tid & 31, w = tid >> 5;
    int wm = (w & 1) * 32, wn = (w >> 1) * 16;
    v8f c0 = {}, c1 = {};
    int r   = tid >> 2;
    int kof = (tid & 3) * 16;
    const __bf16* ga = &A[(long)(bm * 64 + r) * K + kof];
    const __bf16* gb = &Bt[(long)(bn * 64 + r) * K + kof];

    auto issue = [&](int kk, int buf) {
        async_cp16(&As[buf][r * 64 + kof],     ga + kk);
        async_cp16(&As[buf][r * 64 + kof + 8], ga + kk + 8);
        async_cp16(&Bs[buf][r * 64 + kof],     gb + kk);
        async_cp16(&Bs[buf][r * 64 + kof + 8], gb + kk + 8);
    };

    int nk = K >> 6;
    issue(0, 0);
    for (int k = 0; k < nk; ++k) {
        int cur = k & 1;
        if (k + 1 < nk) {
            __syncthreads();               // everyone done reading buffer cur^1
            issue((k + 1) << 6, cur ^ 1);  // prefetch next chunk
            wait_async<4>();               // retire current chunk (in-order)
        } else {
            wait_async<0>();
        }
        __syncthreads();                   // chunk visible to all waves
#pragma unroll
        for (int ks = 0; ks < 64; ks += 32) {
            v16bf a0 = ldfrag(As[cur], wm,      64, ks, lane);
            v16bf a1 = ldfrag(As[cur], wm + 16, 64, ks, lane);
            v16bf b  = ldfrag(Bs[cur], wn,      64, ks, lane);
            c0 = wmma_bf16(a0, b, c0);
            c1 = wmma_bf16(a1, b, c1);
        }
    }
    store_tile(c0, bm * 64 + wm,      bn * 64 + wn, N, Cf, Cb, bias, lane);
    store_tile(c1, bm * 64 + wm + 16, bn * 64 + wn, N, Cf, Cb, bias, lane);
}

// ---------------- GRU gate update ----------------
__device__ __forceinline__ float sigmoidf(float x) { return 1.0f / (1.0f + __expf(-x)); }

__global__ __launch_bounds__(256) void k_gru_update(const float* __restrict__ gi,
                                                    const float* __restrict__ gh,
                                                    const float* __restrict__ hbase, int hstride,
                                                    float* __restrict__ out_states,
                                                    __bf16* __restrict__ hb, int t) {
    int b = blockIdx.x, i = threadIdx.x;
    const float* g  = &gi[(long)(b * LL + t) * 768];
    const float* gg = &gh[b * 768];
    float hp = hbase[(long)b * hstride + i];
    float rr = sigmoidf(g[i]       + gg[i]);
    float zz = sigmoidf(g[256 + i] + gg[256 + i]);
    float nn = tanhf(g[512 + i] + rr * gg[512 + i]);
    float hn = (1.0f - zz) * nn + zz * hp;
    out_states[(long)(b * LL + t) * HH + i] = hn;
    hb[b * HH + i] = (__bf16)hn;
}

// ---------------- attention scores: per-memory dot with its sample's states ----------------
__global__ __launch_bounds__(256) void k_scores(const float* __restrict__ out_states,
                                                const float* __restrict__ std_reps,
                                                const float* __restrict__ copy_reps,
                                                const int* __restrict__ origin,
                                                float* __restrict__ std_scores,
                                                float* __restrict__ copy_scores) {
    int n = blockIdx.x, tid = threadIdx.x;
    __shared__ float sr[HH], cr[HH];
    sr[tid] = std_reps[(long)n * HH + tid];
    cr[tid] = copy_reps[(long)n * HH + tid];
    __syncthreads();
    int l = tid >> 3, j = tid & 7;          // 32 l's x 8 lanes
    int b = origin[n];
    const float* h = &out_states[(long)(b * LL + l) * HH + j * 32];
    float ss = 0.f, cs = 0.f;
#pragma unroll
    for (int k = 0; k < 32; ++k) {
        float hv = h[k];
        ss += hv * sr[j * 32 + k];
        cs += hv * cr[j * 32 + k];
    }
#pragma unroll
    for (int m = 1; m < 8; m <<= 1) {
        ss += __shfl_xor(ss, m, 32);
        cs += __shfl_xor(cs, m, 32);
    }
    if (j == 0) {
        std_scores[n * LL + l]  = ss;
        copy_scores[n * LL + l] = cs;
    }
}

// ---------------- segment softmax + std_out + total_copy + build cat (bf16) ----------------
__global__ __launch_bounds__(256) void k_softmax_stdout(const float* __restrict__ std_scores,
                                                        const float* __restrict__ copy_scores,
                                                        const float* __restrict__ std_reps,
                                                        const float* __restrict__ out_states,
                                                        __bf16* __restrict__ catb,
                                                        float* __restrict__ total_copy) {
    int b = blockIdx.x, tid = threadIdx.x;
    __shared__ float wgt[SEG][LL];
    if (tid < 32) {                       // std softmax weights over the 64-memory segment
        int l = tid;
        float m = NEG_INF;
        for (int s = 0; s < SEG; ++s) m = fmaxf(m, std_scores[(b * SEG + s) * LL + l]);
        float sum = 0.f;
        for (int s = 0; s < SEG; ++s) {
            float e = __expf(std_scores[(b * SEG + s) * LL + l] - m);
            wgt[s][l] = e; sum += e;
        }
        float inv = 1.0f / sum;
        for (int s = 0; s < SEG; ++s) wgt[s][l] *= inv;
    } else if (tid < 64) {                // copy segment logsumexp
        int l = tid - 32;
        float m = NEG_INF;
        for (int s = 0; s < SEG; ++s) m = fmaxf(m, copy_scores[(b * SEG + s) * LL + l]);
        float sum = 0.f;
        for (int s = 0; s < SEG; ++s) sum += __expf(copy_scores[(b * SEG + s) * LL + l] - m);
        total_copy[b * LL + l] = m + __logf(sum);
    }
    __syncthreads();
    float acc[LL];
#pragma unroll
    for (int l = 0; l < LL; ++l) acc[l] = 0.f;
    for (int s = 0; s < SEG; ++s) {
        float rep = std_reps[(long)(b * SEG + s) * HH + tid];
#pragma unroll
        for (int l = 0; l < LL; ++l) acc[l] += wgt[s][l] * rep;
    }
#pragma unroll
    for (int l = 0; l < LL; ++l) {
        long rr = (long)(b * LL + l);
        catb[rr * (2 * HH) + tid]      = (__bf16)acc[l];
        catb[rr * (2 * HH) + HH + tid] = (__bf16)out_states[rr * HH + tid];
    }
}

// ---------------- fused vocab projection + partial logsumexp + token gather ----------------
// 128 rows x 128 vocab cols per block; async double-buffered K staging;
// 64KB LDS reused as the f32 score tile for the in-LDS partial logsumexp.
__global__ __launch_bounds__(256) void k_vocab(const __bf16* __restrict__ Tmp,   // 4096x256
                                               const __bf16* __restrict__ Emb,   // V x 256
                                               const float* __restrict__ vbias,
                                               const int* __restrict__ tok,
                                               float* __restrict__ pmax,
                                               float* __restrict__ psum,
                                               float* __restrict__ tokscore) {
    __shared__ __align__(16) char smem[65536];
    // layout (bytes): As0 @0, As1 @16384, Bs0 @32768, Bs1 @49152; Sc overlays all.
    __bf16* bufp = (__bf16*)smem;
    auto Asb = [&](int b) { return bufp + b * 8192; };
    auto Bsb = [&](int b) { return bufp + 16384 + b * 8192; };
    int br = blockIdx.x;                       // row block: 128 rows
    int bc = blockIdx.y;                       // col chunk: 128 vocab cols
    int tid = threadIdx.x, lane = tid & 31, w = tid >> 5;
    v8f c[8];
#pragma unroll
    for (int i = 0; i < 8; ++i) c[i] = (v8f){};

    // per-thread: 4 uint4 of A + 4 uint4 of B per 64-wide K chunk.
    // Out-of-range emb rows are CLAMPED (uniform ASYNCcnt per wave); their
    // columns are masked with -inf in the epilogue, so garbage is harmless.
    auto issue = [&](int kc, int buf) {
#pragma unroll
        for (int i = 0; i < 4; ++i) {
            int u = tid * 4 + i;
            int row = u >> 3, q = u & 7;
            async_cp16(Asb(buf) + row * 64 + q * 8,
                       &Tmp[(long)(br * 128 + row) * EE + kc * 64 + q * 8]);
            int v = bc * 128 + row;
            int vc = v < VV ? v : (VV - 1);
            async_cp16(Bsb(buf) + row * 64 + q * 8,
                       &Emb[(long)vc * EE + kc * 64 + q * 8]);
        }
    };

    issue(0, 0);
    for (int kc = 0; kc < 4; ++kc) {
        int cur = kc & 1;
        if (kc + 1 < 4) {
            __syncthreads();
            issue(kc + 1, cur ^ 1);
            wait_async<8>();
        } else {
            wait_async<0>();
        }
        __syncthreads();
#pragma unroll
        for (int ks = 0; ks < 64; ks += 32) {
            v16bf a = ldfrag(Asb(cur), w * 16, 64, ks, lane);
#pragma unroll
            for (int ct = 0; ct < 8; ++ct) {
                v16bf bfr = ldfrag(Bsb(cur), ct * 16, 64, ks, lane);
                c[ct] = wmma_bf16(a, bfr, c[ct]);
            }
        }
    }
    __syncthreads();
    // dump scores (+bias, invalid cols = -inf) to LDS: float[128][128] overlays all buffers
    float* Sc = (float*)smem;
    int cn = lane & 15, rb = (lane >> 4) << 3;
#pragma unroll
    for (int ct = 0; ct < 8; ++ct) {
        int col  = ct * 16 + cn;
        int gcol = bc * 128 + col;
        float bv = (gcol < VV) ? vbias[gcol] : 0.f;
#pragma unroll
        for (int j = 0; j < 8; ++j) {
            int row = w * 16 + rb + j;
            Sc[row * 128 + col] = (gcol < VV) ? (c[ct][j] + bv) : NEG_INF;
        }
    }
    __syncthreads();
    // per-row partial (max, sum exp) over this 128-col chunk; 2 threads per row
    int row = tid >> 1, half = tid & 1;
    float m = NEG_INF;
    for (int k = 0; k < 64; ++k) m = fmaxf(m, Sc[row * 128 + half * 64 + k]);
    float m2 = fmaxf(m, __shfl_xor(m, 1, 32));
    float s = 0.f;
    for (int k = 0; k < 64; ++k) s += __expf(Sc[row * 128 + half * 64 + k] - m2);
    float s2 = s + __shfl_xor(s, 1, 32);
    if (half == 0) {
        int gr = br * 128 + row;
        pmax[(long)gr * NCHUNK + bc] = m2;
        psum[(long)gr * NCHUNK + bc] = s2;
        int b = gr >> 5, l = gr & 31;
        int tk = tok[b * TT + l + 1];          // tok_out
        int rel = tk - bc * 128;
        if (rel >= 0 && rel < 128) tokscore[gr] = Sc[row * 128 + rel];
    }
}

// ---------------- combine partials + total_copy -> norm, gen logprob at target ----------------
__global__ void k_norm(const float* __restrict__ pmax, const float* __restrict__ psum,
                       const float* __restrict__ tokscore, const float* __restrict__ total_copy,
                       float* __restrict__ normv, float* __restrict__ genlp) {
    int r = blockIdx.x * blockDim.x + threadIdx.x;
    if (r >= BL) return;
    float M = NEG_INF;
    for (int k = 0; k < NCHUNK; ++k) M = fmaxf(M, pmax[(long)r * NCHUNK + k]);
    float S = 0.f;
    for (int k = 0; k < NCHUNK; ++k)
        S += psum[(long)r * NCHUNK + k] * __expf(pmax[(long)r * NCHUNK + k] - M);
    float tc = total_copy[r];
    float M2 = fmaxf(M, tc);
    float S2 = S * __expf(M - M2) + __expf(tc - M2);
    float nrm = M2 + __logf(S2);
    normv[r] = nrm;
    genlp[r] = tokscore[r] - nrm;
}

// ---------------- copy logsumexp over irregular segments (atomic 3-pass) ----------------
__device__ __forceinline__ unsigned f2ord(float f) {
    unsigned u = __float_as_uint(f);
    return (u & 0x80000000u) ? ~u : (u | 0x80000000u);
}
__device__ __forceinline__ float ord2f(unsigned u) {
    unsigned b = (u & 0x80000000u) ? (u ^ 0x80000000u) : ~u;
    return __uint_as_float(b);
}

__global__ void k_cc_init(unsigned* cmax, float* csum, int* ccnt) {
    int i = blockIdx.x * blockDim.x + threadIdx.x;
    if (i < BL) { cmax[i] = 0u; csum[i] = 0.f; ccnt[i] = 0; }
}

__device__ __forceinline__ float copy_lp_at(int src, const float* copy_scores,
                                            const float* normv, const int* origin) {
    int n = src >> 5, l = src & 31;
    int b = origin[n];
    return copy_scores[src] - normv[b * LL + l];
}

__global__ void k_cc_pass1(const int* __restrict__ eidx, const int* __restrict__ sidx,
                           const float* __restrict__ copy_scores, const float* __restrict__ normv,
                           const int* __restrict__ origin, unsigned* cmax, int* ccnt) {
    int ci = blockIdx.x * blockDim.x + threadIdx.x;
    if (ci >= CC) return;
    float val = copy_lp_at(eidx[ci], copy_scores, normv, origin);
    int t = sidx[ci];
    atomicMax(&cmax[t], f2ord(val));
    atomicAdd(&ccnt[t], 1);
}

__global__ void k_cc_pass2(const int* __restrict__ eidx, const int* __restrict__ sidx,
                           const float* __restrict__ copy_scores, const float* __restrict__ normv,
                           const int* __restrict__ origin, const unsigned* cmax, float* csum) {
    int ci = blockIdx.x * blockDim.x + threadIdx.x;
    if (ci >= CC) return;
    float val = copy_lp_at(eidx[ci], copy_scores, normv, origin);
    int t = sidx[ci];
    atomicAdd(&csum[t], __expf(val - ord2f(cmax[t])));
}

// ---------------- final loss reduction ----------------
__global__ __launch_bounds__(128) void k_loss(const float* __restrict__ genlp,
                                              const unsigned* __restrict__ cmax,
                                              const float* __restrict__ csum,
                                              const int* __restrict__ ccnt,
                                              const int* __restrict__ tok,
                                              const int* __restrict__ tlen,
                                              float* __restrict__ out) {
    __shared__ float red[128];
    int b = threadIdx.x;
    int len = tlen[b];
    float acc = 0.f;
    for (int l = 0; l < LL; ++l) {
        int r = b * LL + l;
        int cnt = ccnt[r];
        float cc = (cnt > 0) ? (ord2f(cmax[r]) + __logf(csum[r])) : NEG_INF;
        int tk = tok[b * TT + l + 1];
        bool mask = (cnt > 0) && (tk == 1);            // UNK_ID == 1
        float g = mask ? NEG_INF : genlp[r];
        float m = fmaxf(g, cc);
        float any = isinf(m) && m < 0.f ? NEG_INF
                                        : m + __logf(__expf(g - m) + __expf(cc - m));
        if (l < len) acc += any;
    }
    red[b] = acc / (float)len;
    __syncthreads();
    for (int s = 64; s > 0; s >>= 1) {
        if (b < s) red[b] += red[b + s];
        __syncthreads();
    }
    if (b == 0) out[0] = -red[0] / (float)BB;
}

// ---------------- host orchestration ----------------
extern "C" void kernel_launch(void* const* d_in, const int* in_sizes, int n_in,
                              void* d_out, int out_size, void* d_ws, size_t ws_size,
                              hipStream_t stream) {
    (void)in_sizes; (void)n_in; (void)out_size; (void)ws_size;
    const float* in_mem   = (const float*)d_in[0];
    const int*   origin   = (const int*)  d_in[1];
    const float* init_st  = (const float*)d_in[2];
    const int*   tok      = (const int*)  d_in[3];
    const int*   eidx     = (const int*)  d_in[4];
    const int*   sidx     = (const int*)  d_in[5];
    const int*   tlen     = (const int*)  d_in[6];
    const float* emb      = (const float*)d_in[7];
    const float* w_ih     = (const float*)d_in[8];
    const float* w_hh     = (const float*)d_in[9];
    const float* b_ih     = (const float*)d_in[10];
    const float* b_hh     = (const float*)d_in[11];
    const float* w_std    = (const float*)d_in[12];
    const float* w_copy   = (const float*)d_in[13];
    const float* h2v      = (const float*)d_in[14];
    const float* vbias    = (const float*)d_in[15];

    size_t off = 0;
    auto alloc = [&](size_t bytes) -> void* {
        void* p = (char*)d_ws + off;
        off += (bytes + 255) & ~(size_t)255;
        return p;
    };
    __bf16* embb    = (__bf16*)alloc((size_t)VV * EE * 2);
    __bf16* memb    = (__bf16*)alloc((size_t)NN * DD * 2);
    __bf16* xb      = (__bf16*)alloc((size_t)BL * EE * 2);
    __bf16* wihb    = (__bf16*)alloc((size_t)768 * EE * 2);
    __bf16* whhb    = (__bf16*)alloc((size_t)768 * HH * 2);
    __bf16* wstdb   = (__bf16*)alloc((size_t)HH * DD * 2);
    __bf16* wcopyb  = (__bf16*)alloc((size_t)HH * DD * 2);
    __bf16* h2vTb   = (__bf16*)alloc((size_t)EE * (2 * HH) * 2);
    float*  gi      = (float*) alloc((size_t)BL * 768 * 4);
    float*  ghbuf   = (float*) alloc((size_t)BB * 768 * 4);
    __bf16* hb      = (__bf16*)alloc((size_t)BB * HH * 2);
    float*  ostates = (float*) alloc((size_t)BL * HH * 4);
    float*  sreps   = (float*) alloc((size_t)NN * HH * 4);
    float*  creps   = (float*) alloc((size_t)NN * HH * 4);
    float*  sscore  = (float*) alloc((size_t)NN * LL * 4);
    float*  cscore  = (float*) alloc((size_t)NN * LL * 4);
    float*  tcopy   = (float*) alloc((size_t)BL * 4);
    __bf16* catb    = (__bf16*)alloc((size_t)BL * (2 * HH) * 2);
    __bf16* tmpb    = (__bf16*)alloc((size_t)BL * EE * 2);
    float*  pmax    = (float*) alloc((size_t)BL * NCHUNK * 4);
    float*  psum    = (float*) alloc((size_t)BL * NCHUNK * 4);
    float*  tokscr  = (float*) alloc((size_t)BL * 4);
    float*  normv   = (float*) alloc((size_t)BL * 4);
    float*  genlp   = (float*) alloc((size_t)BL * 4);
    unsigned* cmax  = (unsigned*)alloc((size_t)BL * 4);
    float*  csum    = (float*) alloc((size_t)BL * 4);
    int*    ccnt    = (int*)   alloc((size_t)BL * 4);

    auto cvt = [&](const float* src, __bf16* dst, int n) {
        k_cvt_bf16<<<(n + 255) / 256, 256, 0, stream>>>(src, dst, n);
    };
    // 1) precision converts
    cvt(emb,    embb,   VV * EE);
    cvt(in_mem, memb,   NN * DD);
    cvt(w_ih,   wihb,   768 * EE);
    cvt(w_hh,   whhb,   768 * HH);
    cvt(w_std,  wstdb,  HH * DD);
    cvt(w_copy, wcopyb, HH * DD);
    k_cvt_t_bf16<<<(512 * 256 + 255) / 256, 256, 0, stream>>>(h2v, h2vTb, 2 * HH, EE);

    // 2) embedding gather + gi = X @ W_ih^T + b_ih
    k_xgather<<<BL, 256, 0, stream>>>(emb, tok, xb);
    k_gemm_bf16<<<dim3(BL / 64, 768 / 64), 256, 0, stream>>>(xb, wihb, gi, nullptr, b_ih,
                                                             BL, 768, EE);
    // 3) GRU recurrence
    cvt(init_st, hb, BB * HH);
    for (int t = 0; t < LL; ++t) {
        k_gemm_bf16<<<dim3(BB / 64, 768 / 64), 256, 0, stream>>>(hb, whhb, ghbuf, nullptr, b_hh,
                                                                 BB, 768, HH);
        const float* hbase = (t == 0) ? init_st : (ostates + (size_t)(t - 1) * HH);
        int hstride = (t == 0) ? HH : (LL * HH);
        k_gru_update<<<BB, HH, 0, stream>>>(gi, ghbuf, hbase, hstride, ostates, hb, t);
    }
    // 4) attention reps
    k_gemm_bf16<<<dim3(NN / 64, HH / 64), 256, 0, stream>>>(memb, wstdb,  sreps, nullptr, nullptr,
                                                            NN, HH, DD);
    k_gemm_bf16<<<dim3(NN / 64, HH / 64), 256, 0, stream>>>(memb, wcopyb, creps, nullptr, nullptr,
                                                            NN, HH, DD);
    // 5) scores + segment softmax / std_out / total_copy, build cat (bf16)
    k_scores<<<NN, 256, 0, stream>>>(ostates, sreps, creps, origin, sscore, cscore);
    k_softmax_stdout<<<BB, 256, 0, stream>>>(sscore, cscore, sreps, ostates, catb, tcopy);
    // 6) tmp = cat @ h2v  (bf16 out)
    k_gemm_bf16<<<dim3(BL / 64, EE / 64), 256, 0, stream>>>(catb, h2vTb, nullptr, tmpb, nullptr,
                                                            BL, EE, 2 * HH);
    // 7) fused vocab projection + partial logsumexp + token gather
    k_vocab<<<dim3(BL / 128, NCHUNK), 256, 0, stream>>>(tmpb, embb, vbias, tok,
                                                        pmax, psum, tokscr);
    // 8) norm + gen logprob
    k_norm<<<(BL + 255) / 256, 256, 0, stream>>>(pmax, psum, tokscr, tcopy, normv, genlp);
    // 9) copy logsumexp (atomic 3-pass)
    k_cc_init<<<(BL + 255) / 256, 256, 0, stream>>>(cmax, csum, ccnt);
    k_cc_pass1<<<(CC + 255) / 256, 256, 0, stream>>>(eidx, sidx, cscore, normv, origin, cmax, ccnt);
    k_cc_pass2<<<(CC + 255) / 256, 256, 0, stream>>>(eidx, sidx, cscore, normv, origin, cmax, csum);
    // 10) final loss
    k_loss<<<1, 128, 0, stream>>>(genlp, cmax, csum, ccnt, tok, tlen, (float*)d_out);
}
